// BEVLifter_46239617909270
// MI455X (gfx1250) — matrix operand order repaired and checked by reading, not compiled, not run on hardware
//
#include <hip/hip_runtime.h>
#include <hip/hip_bf16.h>

typedef __attribute__((ext_vector_type(8)))  _Float16 v8h;
typedef __attribute__((ext_vector_type(16))) _Float16 v16h;
typedef __attribute__((ext_vector_type(8)))  float    v8f;
typedef __attribute__((ext_vector_type(4)))  unsigned int v4u;
typedef __attribute__((ext_vector_type(8)))  int      v8i;
typedef __attribute__((ext_vector_type(4)))  int      v4i;

#define DEV __device__ __forceinline__

constexpr int   BB    = 4;
constexpr int   INCH  = 256;
constexpr int   HF    = 40;
constexpr int   WFp   = 100;
constexpr int   Cc    = 64;
constexpr int   Dd    = 48;
constexpr int   NPIX  = HF * WFp;     // 4000
constexpr int   NXc   = 40;
constexpr int   NZc   = 99;
constexpr int   GRIDC = NZc * NXc;    // 3960
constexpr int   K33   = Cc * 9;       // 576
constexpr float X0f   = -10.0f;
constexpr float Z0f   = 0.5f;
constexpr float RESf  = 0.5f;
constexpr float DMINf = 0.5f;
constexpr float DMAXf = 50.0f;
constexpr float EPSf  = 1e-5f;

DEV v8f wmma_f16(v16h a, v16h b, v8f c) {
  return __builtin_amdgcn_wmma_f32_16x16x32_f16(
      false, a, false, b, (short)0, c, false, false);
}

DEV v16h cat8(v8h lo, v8h hi) {
  return __builtin_shufflevector(lo, hi, 0, 1, 2, 3, 4, 5, 6, 7,
                                 8, 9, 10, 11, 12, 13, 14, 15);
}

// ---------------------------------------------------------------------------
// TDM: DMA a 1-D run of `ndwords` DWORDs from global into LDS.
// D# built per CDNA5 ISA §8: group0 {count=1, lds_addr, global_addr, type=2},
// group1 {data_size=4B, tensor_dim0=tile_dim0=ndwords, stride=ndwords},
// groups 2/3 (and the trailing group on this toolchain) zero = dims unused.
// Issue from ONE wave (EXEC ignored by TDM), wait TENSORcnt, then barrier.
// ---------------------------------------------------------------------------
DEV void tdm_stage_to_lds(const void* gsrc, void* lds_dst, int ndwords) {
  const unsigned long long ga = (unsigned long long)(uintptr_t)gsrc;
  const unsigned ldsoff = (unsigned)(uintptr_t)lds_dst;  // LDS byte offset (addr[31:0])
  v4u g0;
  g0[0] = 1u;                                            // count=1, user descriptor
  g0[1] = ldsoff;                                        // lds_addr [63:32]
  g0[2] = (unsigned)ga;                                  // global_addr[31:0]
  g0[3] = (unsigned)((ga >> 32) & 0x01FFFFFFu) | (2u << 30); // addr[56:32] | type=2
  const unsigned nd = (unsigned)ndwords;
  v8i g1;
  g1[0] = (int)(2u << 16);                               // data_size = 4 bytes
  g1[1] = (int)((nd & 0xFFFFu) << 16);                   // tensor_dim0[15:0]  (bits 63:48)
  g1[2] = (int)(((nd >> 16) & 0xFFFFu) | (1u << 16));    // tensor_dim0[31:16], tensor_dim1=1
  g1[3] = (int)((nd & 0xFFFFu) << 16);                   // tile_dim0 (bits 127:112)
  g1[4] = 0;                                             // tile_dim1=0, tile_dim2=0 (unused)
  g1[5] = (int)nd;                                       // tensor_dim0_stride[31:0]
  g1[6] = 0;
  g1[7] = 0;
  v4i g2 = (v4i)0;
  v4i g3 = (v4i)0;
  v8i g4 = (v8i)0;                                       // clang-23 6-arg form
  __builtin_amdgcn_tensor_load_to_lds(g0, g1, g2, g3, g4, 0);
}

// ---------------------------------------------------------------------------
// channel-last f16 GEMM:  out[b,n,m] = relu?( scale[m]*(A[m,:]·X[b,n,:]) + shift[m] )
// A (M,K) f16 row-major, TDM-staged whole into LDS.  X (B,N,K) f16.
// block = one (b, n-tile); wave w = m-tile w.  M,N mult of 16, K mult of 32.
// ---------------------------------------------------------------------------
__global__ void k_gemm_cl(const _Float16* __restrict__ A16,   // (M,K)
                          const _Float16* __restrict__ X16,   // (B,N,K)
                          const float* __restrict__ scale,
                          const float* __restrict__ shift,
                          _Float16* __restrict__ out16,       // (B,N,M)
                          int M, int K, int N, int relu)
{
  extern __shared__ _Float16 As[];
  if (threadIdx.x < 32) {                       // one wave issues the TDM DMA
    tdm_stage_to_lds(A16, As, (M * K) >> 1);    // halfs -> dwords
    __builtin_amdgcn_s_wait_tensorcnt(0);
  }
  __syncthreads();

  const int lane = threadIdx.x & 31;
  const int m0   = (threadIdx.x >> 5) << 4;
  const int nt   = N >> 4;
  const int b    = blockIdx.x / nt;
  const int n0   = (blockIdx.x % nt) << 4;

  const int ml = m0 + (lane & 15);
  const int nl = n0 + (lane & 15);
  const int ka_off = (lane & 16) ? 16 : 0;       // contiguous k = 16h + s
  const int kb_off = (lane & 16) ? 16 : 0;       // contiguous k = 16h + s, same as A
  const _Float16* Ar = As + ml * K;
  const _Float16* Xr = X16 + ((size_t)b * N + nl) * K;

  v8f acc = {};
  for (int k0 = 0; k0 < K; k0 += 32) {
    v8h a0 = *(const v8h*)(Ar + k0 + ka_off);         // ds_load_b128
    v8h a1 = *(const v8h*)(Ar + k0 + ka_off + 8);
    v8h b0 = *(const v8h*)(Xr + k0 + kb_off);         // global_load_b128
    v8h b1 = *(const v8h*)(Xr + k0 + kb_off + 8);
    acc = wmma_f16(cat8(a0, a1), cat8(b0, b1), acc);
  }

  const int mo = m0 + ((lane & 16) ? 8 : 0);
  v8h ov;
#pragma unroll
  for (int r = 0; r < 8; ++r) {
    float v = acc[r] * scale[mo + r] + shift[mo + r];
    if (relu) v = fmaxf(v, 0.0f);
    ov[r] = (_Float16)v;
  }
  *(v8h*)(out16 + ((size_t)b * N + nl) * M + mo) = ov;  // global_store_b128
}

// ---------------------------------------------------------------------------
// 3x3 conv (pad 1), implicit GEMM, channel-last.
// A16 pre-shuffled to (Cout=64, 9, Cin=64) so K order is k = r9*64 + cin:
// every 16-half fragment group = one shifted pixel, contiguous channels.
// X (B,Nn,64) f16.  Output: f16 channel-last, or f32 channel-first (last layer).
// ---------------------------------------------------------------------------
__global__ void k_conv3_cl(const _Float16* __restrict__ A16,  // (64, 576)
                           const _Float16* __restrict__ X16,  // (B, Nn, 64)
                           const float* __restrict__ scale,
                           const float* __restrict__ shift,
                           _Float16* __restrict__ out16,      // (B, Nn, 64) or null
                           float* __restrict__ outf,          // (B, 64, Nn) or null
                           int Hc, int Wc, int chfirst)
{
  const int Nn = Hc * Wc;
  extern __shared__ _Float16 As[];
  if (threadIdx.x < 32) {
    tdm_stage_to_lds(A16, As, (Cc * K33) >> 1);
    __builtin_amdgcn_s_wait_tensorcnt(0);
  }
  __syncthreads();

  const int lane = threadIdx.x & 31;
  const int m0   = (threadIdx.x >> 5) << 4;
  const int nt   = (Nn + 15) >> 4;
  const int b    = blockIdx.x / nt;
  const int n0   = (blockIdx.x % nt) << 4;

  const int ml = m0 + (lane & 15);
  const int nl = n0 + (lane & 15);
  const bool nok = (nl < Nn);
  const int y = nok ? (nl / Wc) : 0;
  const int x = nok ? (nl % Wc) : 0;
  const int ka_off = (lane & 16) ? 16 : 0;
  const int kb_off = (lane & 16) ? 16 : 0;
  const _Float16* Ar = As + ml * K33;
  const _Float16* Xb = X16 + (size_t)b * Nn * Cc;

  v8f acc = {};
  for (int k0 = 0; k0 < K33; k0 += 32) {
    const int kk   = k0 + kb_off;        // 16-aligned, stays inside one 64-block
    const int r9   = kk >> 6;            // tap index 0..8
    const int cin0 = kk & 63;            // 0/16/32/48
    const int yy = y + r9 / 3 - 1;
    const int xx = x + r9 % 3 - 1;
    v8h b0 = {}, b1 = {};
    if (nok && yy >= 0 && yy < Hc && xx >= 0 && xx < Wc) {
      const _Float16* p = Xb + ((size_t)(yy * Wc + xx)) * Cc + cin0;
      b0 = *(const v8h*)p;               // global_load_b128
      b1 = *(const v8h*)(p + 8);
    }
    v8h a0 = *(const v8h*)(Ar + k0 + ka_off);        // ds_load_b128
    v8h a1 = *(const v8h*)(Ar + k0 + ka_off + 8);
    acc = wmma_f16(cat8(a0, a1), cat8(b0, b1), acc); // EXEC all-ones here
  }

  const int mo = m0 + ((lane & 16) ? 8 : 0);
  if (chfirst) {
#pragma unroll
    for (int r = 0; r < 8; ++r) {
      float v = fmaxf(acc[r] * scale[mo + r] + shift[mo + r], 0.0f);
      if (nok) outf[((size_t)b * Cc + mo + r) * Nn + nl] = v;
    }
  } else {
    v8h ov;
#pragma unroll
    for (int r = 0; r < 8; ++r) {
      float v = fmaxf(acc[r] * scale[mo + r] + shift[mo + r], 0.0f);
      ov[r] = (_Float16)v;
    }
    if (nok) *(v8h*)(out16 + ((size_t)b * Nn + nl) * Cc + mo) = ov;
  }
}

// ---------------------------------------------------------------------------
// prep kernels
// ---------------------------------------------------------------------------
__global__ void k_bnprep(const float* __restrict__ bias, const float* __restrict__ g,
                         const float* __restrict__ be, const float* __restrict__ mu,
                         const float* __restrict__ var,
                         float* __restrict__ scale, float* __restrict__ shift,
                         int M, int has_bn)
{
  const int m = blockIdx.x * blockDim.x + threadIdx.x;
  if (m >= M) return;
  if (has_bn) {
    const float s = g[m] * rsqrtf(var[m] + EPSf);
    scale[m] = s;
    shift[m] = (bias[m] - mu[m]) * s + be[m];
  } else {
    scale[m] = 1.0f;
    shift[m] = bias[m];
  }
}

__global__ void k_cvt16(const float* __restrict__ in, _Float16* __restrict__ out, int n)
{
  const int i = blockIdx.x * blockDim.x + threadIdx.x;
  if (i < n) out[i] = (_Float16)in[i];
}

// (O, I, 3, 3) f32 -> (O, 9, I) f16
__global__ void k_shw3(const float* __restrict__ w, _Float16* __restrict__ out)
{
  const int t = blockIdx.x * blockDim.x + threadIdx.x;
  if (t >= Cc * K33) return;
  const int o = t / K33, rem = t % K33;
  const int r9 = rem / Cc, cin = rem % Cc;
  out[t] = (_Float16)w[(size_t)o * K33 + cin * 9 + r9];
}

// (B, 256, N) f32 -> (B, N, 256) f16
__global__ void k_trfeats(const float* __restrict__ in, _Float16* __restrict__ out)
{
  const int t = blockIdx.x * blockDim.x + threadIdx.x;
  if (t >= BB * INCH * NPIX) return;
  const int b = t / (INCH * NPIX), rem = t % (INCH * NPIX);
  const int k = rem / NPIX, n = rem % NPIX;
  out[((size_t)b * NPIX + n) * INCH + k] = (_Float16)in[t];
}

__global__ void k_softmax_cl(_Float16* __restrict__ pr)   // (B*N, D) contiguous
{
  const int t = blockIdx.x * blockDim.x + threadIdx.x;
  if (t >= BB * NPIX) return;
  _Float16* pp = pr + (size_t)t * Dd;
  float v[Dd];
  float mx = -3.4e38f;
#pragma unroll
  for (int d = 0; d < Dd; ++d) { v[d] = (float)pp[d]; mx = fmaxf(mx, v[d]); }
  float s = 0.0f;
#pragma unroll
  for (int d = 0; d < Dd; ++d) { v[d] = expf(v[d] - mx); s += v[d]; }
  const float inv = 1.0f / s;
#pragma unroll
  for (int d = 0; d < Dd; ++d) pp[d] = (_Float16)(v[d] * inv);
}

__global__ void k_zero(float* __restrict__ p, int n)
{
  const int i = blockIdx.x * blockDim.x + threadIdx.x;
  if (i < n) p[i] = 0.0f;
}

// ---------------------------------------------------------------------------
// rays + binning + scatter-add; channel-last layouts, contiguous per cell
// ---------------------------------------------------------------------------
__global__ void k_splat_cl(const float* __restrict__ Kmat,
                           const int*   __restrict__ Hp,
                           const int*   __restrict__ Wp,
                           const _Float16* __restrict__ ctx,   // (B,N,64)
                           const _Float16* __restrict__ probs, // (B,N,48)
                           float* __restrict__ bev)            // (B,GRID,64)
{
  const int t = blockIdx.x * blockDim.x + threadIdx.x;
  if (t >= BB * NPIX) return;
  const int b = t / NPIX, n = t % NPIX;

  const float sx = (float)WFp / (float)Wp[0];
  const float sy = (float)HF  / (float)Hp[0];
  const float* Kb = Kmat + b * 9;
  const float a00 = Kb[0] * sx, a01 = Kb[1] * sx, a02 = Kb[2] * sx;
  const float a10 = Kb[3] * sy, a11 = Kb[4] * sy, a12 = Kb[5] * sy;
  const float a20 = Kb[6],      a21 = Kb[7],      a22 = Kb[8];
  const float det = a00 * (a11 * a22 - a12 * a21)
                  - a01 * (a10 * a22 - a12 * a20)
                  + a02 * (a10 * a21 - a11 * a20);
  const float id  = 1.0f / det;
  const float i00 = (a11 * a22 - a12 * a21) * id;
  const float i01 = (a02 * a21 - a01 * a22) * id;
  const float i02 = (a01 * a12 - a02 * a11) * id;
  const float i20 = (a10 * a21 - a11 * a20) * id;
  const float i21 = (a01 * a20 - a00 * a21) * id;
  const float i22 = (a00 * a11 - a01 * a10) * id;

  const float u = (float)(n % WFp);
  const float v = (float)(n / WFp);
  const float rx = i00 * u + i01 * v + i02;
  const float rz = i20 * u + i21 * v + i22;

  const _Float16* cb = ctx   + ((size_t)b * NPIX + n) * Cc;
  const _Float16* pb = probs + ((size_t)b * NPIX + n) * Dd;
  float cv[Cc];
#pragma unroll
  for (int c = 0; c < Cc; ++c) cv[c] = (float)cb[c];

  float* gbase = bev + (size_t)b * GRIDC * Cc;
  for (int d = 0; d < Dd; ++d) {
    const float dc = DMINf + (DMAXf - DMINf) * (float)d / (float)(Dd - 1);
    const int bx = (int)((rx * dc - X0f) / RESf);   // trunc toward 0 == ref
    const int bz = (int)((rz * dc - Z0f) / RESf);
    if (bx < 0 || bx >= NXc || bz < 0 || bz >= NZc) continue;
    float* gb = gbase + (size_t)(bz * NXc + bx) * Cc;
    const float p = (float)pb[d];
    for (int c = 0; c < Cc; ++c)
      atomicAdd(gb + c, cv[c] * p);
  }
}

// ---------------------------------------------------------------------------
extern "C" void kernel_launch(void* const* d_in, const int* in_sizes, int n_in,
                              void* d_out, int out_size, void* d_ws, size_t ws_size,
                              hipStream_t stream)
{
  (void)in_sizes; (void)n_in; (void)out_size; (void)ws_size;

  const float* feats  = (const float*)d_in[0];
  const float* Kmat   = (const float*)d_in[1];
  const int*   Hp     = (const int*)  d_in[2];
  const int*   Wp     = (const int*)  d_in[3];
  const float* w_red  = (const float*)d_in[4];
  const float* b_red  = (const float*)d_in[5];
  const float* g_red  = (const float*)d_in[6];
  const float* be_red = (const float*)d_in[7];
  const float* m_red  = (const float*)d_in[8];
  const float* v_red  = (const float*)d_in[9];
  const float* w_dp1  = (const float*)d_in[10];
  const float* b_dp1  = (const float*)d_in[11];
  const float* g_dp1  = (const float*)d_in[12];
  const float* be_dp1 = (const float*)d_in[13];
  const float* m_dp1  = (const float*)d_in[14];
  const float* v_dp1  = (const float*)d_in[15];
  const float* w_dp2  = (const float*)d_in[16];
  const float* b_dp2  = (const float*)d_in[17];
  const float* w_be1  = (const float*)d_in[18];
  const float* b_be1  = (const float*)d_in[19];
  const float* g_be1  = (const float*)d_in[20];
  const float* be_be1 = (const float*)d_in[21];
  const float* m_be1  = (const float*)d_in[22];
  const float* v_be1  = (const float*)d_in[23];
  const float* w_be2  = (const float*)d_in[24];
  const float* b_be2  = (const float*)d_in[25];
  const float* g_be2  = (const float*)d_in[26];
  const float* be_be2 = (const float*)d_in[27];
  const float* m_be2  = (const float*)d_in[28];
  const float* v_be2  = (const float*)d_in[29];

  // ---- workspace carve (256B aligned chunks) ----
  char* wp = (char*)d_ws;
  auto take = [&](size_t bytes) -> char* {
    char* r = wp;
    wp += (bytes + 255) & ~(size_t)255;
    return r;
  };
  _Float16* ft16   = (_Float16*)take((size_t)BB * NPIX * INCH * 2);
  _Float16* ctx16  = (_Float16*)take((size_t)BB * NPIX * Cc * 2);
  _Float16* h16    = (_Float16*)take((size_t)BB * NPIX * Cc * 2);
  _Float16* pr16   = (_Float16*)take((size_t)BB * NPIX * Dd * 2);
  _Float16* bev16  = (_Float16*)take((size_t)BB * GRIDC * Cc * 2);
  _Float16* bev116 = (_Float16*)take((size_t)BB * GRIDC * Cc * 2);
  float*    bev0f  = (float*)   take((size_t)BB * GRIDC * Cc * 4);
  _Float16* wred16 = (_Float16*)take((size_t)Cc * INCH * 2);
  _Float16* wdp1s  = (_Float16*)take((size_t)Cc * K33 * 2);
  _Float16* wdp216 = (_Float16*)take((size_t)Dd * Cc * 2);
  _Float16* wbe1s  = (_Float16*)take((size_t)Cc * K33 * 2);
  _Float16* wbe2s  = (_Float16*)take((size_t)Cc * K33 * 2);
  float* sc_red = (float*)take(Cc * 4); float* sh_red = (float*)take(Cc * 4);
  float* sc_dp1 = (float*)take(Cc * 4); float* sh_dp1 = (float*)take(Cc * 4);
  float* sc_dp2 = (float*)take(Dd * 4); float* sh_dp2 = (float*)take(Dd * 4);
  float* sc_be1 = (float*)take(Cc * 4); float* sh_be1 = (float*)take(Cc * 4);
  float* sc_be2 = (float*)take(Cc * 4); float* sh_be2 = (float*)take(Cc * 4);

  const int TPB = 256;

  // ---- prep: layout transforms, f16 conversion, BN folding ----
  k_trfeats<<<(BB * INCH * NPIX + TPB - 1) / TPB, TPB, 0, stream>>>(feats, ft16);
  k_cvt16<<<(Cc * INCH + TPB - 1) / TPB, TPB, 0, stream>>>(w_red, wred16, Cc * INCH);
  k_cvt16<<<(Dd * Cc + TPB - 1) / TPB, TPB, 0, stream>>>(w_dp2, wdp216, Dd * Cc);
  k_shw3<<<(Cc * K33 + TPB - 1) / TPB, TPB, 0, stream>>>(w_dp1, wdp1s);
  k_shw3<<<(Cc * K33 + TPB - 1) / TPB, TPB, 0, stream>>>(w_be1, wbe1s);
  k_shw3<<<(Cc * K33 + TPB - 1) / TPB, TPB, 0, stream>>>(w_be2, wbe2s);
  k_bnprep<<<1, 64, 0, stream>>>(b_red, g_red, be_red, m_red, v_red, sc_red, sh_red, Cc, 1);
  k_bnprep<<<1, 64, 0, stream>>>(b_dp1, g_dp1, be_dp1, m_dp1, v_dp1, sc_dp1, sh_dp1, Cc, 1);
  k_bnprep<<<1, 64, 0, stream>>>(b_dp2, nullptr, nullptr, nullptr, nullptr, sc_dp2, sh_dp2, Dd, 0);
  k_bnprep<<<1, 64, 0, stream>>>(b_be1, g_be1, be_be1, m_be1, v_be1, sc_be1, sh_be1, Cc, 1);
  k_bnprep<<<1, 64, 0, stream>>>(b_be2, g_be2, be_be2, m_be2, v_be2, sc_be2, sh_be2, Cc, 1);

  // 1) context = relu(bn(1x1 conv)) : GEMM 64x256 @ (B,N,256)
  k_gemm_cl<<<BB * (NPIX / 16), 32 * (Cc / 16), Cc * INCH * 2, stream>>>(
      wred16, ft16, sc_red, sh_red, ctx16, Cc, INCH, NPIX, 1);
  // 2) h = relu(bn(3x3 conv)) : implicit GEMM K=576
  k_conv3_cl<<<BB * ((NPIX + 15) / 16), 128, Cc * K33 * 2, stream>>>(
      wdp1s, ctx16, sc_dp1, sh_dp1, h16, nullptr, HF, WFp, 0);
  // 3) depth logits : GEMM 48x64
  k_gemm_cl<<<BB * (NPIX / 16), 32 * (Dd / 16), Dd * Cc * 2, stream>>>(
      wdp216, h16, sc_dp2, sh_dp2, pr16, Dd, Cc, NPIX, 0);
  // 4) softmax over D
  k_softmax_cl<<<(BB * NPIX + TPB - 1) / TPB, TPB, 0, stream>>>(pr16);
  // 5) zero BEV accumulator (fresh every call — graph-replay safe)
  k_zero<<<(BB * GRIDC * Cc + TPB - 1) / TPB, TPB, 0, stream>>>(bev0f, BB * GRIDC * Cc);
  // 6) lift-splat scatter-add
  k_splat_cl<<<(BB * NPIX + TPB - 1) / TPB, TPB, 0, stream>>>(
      Kmat, Hp, Wp, ctx16, pr16, bev0f);
  // 7) BEV accumulator -> f16 channel-last
  k_cvt16<<<(BB * GRIDC * Cc + TPB - 1) / TPB, TPB, 0, stream>>>(
      bev0f, bev16, BB * GRIDC * Cc);
  // 8) bev = relu(bn(3x3 conv))
  k_conv3_cl<<<BB * ((GRIDC + 15) / 16), 128, Cc * K33 * 2, stream>>>(
      wbe1s, bev16, sc_be1, sh_be1, bev116, nullptr, NZc, NXc, 0);
  // 9) final 3x3 conv -> f32 channel-first output
  k_conv3_cl<<<BB * ((GRIDC + 15) / 16), 128, Cc * K33 * 2, stream>>>(
      wbe2s, bev116, sc_be2, sh_be2, nullptr, (float*)d_out, NZc, NXc, 1);
}
